// MIM_18270790877927
// MI455X (gfx1250) — matrix operand order repaired and checked by reading, not compiled, not run on hardware
//
#include <hip/hip_runtime.h>

#define BB   4
#define TT   10
#define HH   64
#define WW   64
#define PIX  4096                 // HH*WW
#define S64  (BB * 64 * PIX)      // elements of a (B,64,H,W) tensor
#define CH64STR (64 * PIX)
#define NPIX16  (BB * PIX / 16)   // 1024 pixel tiles of 16

typedef __attribute__((ext_vector_type(16))) __bf16 v16bf;
typedef __attribute__((ext_vector_type(8)))  float  v8f;
typedef __attribute__((ext_vector_type(4)))  unsigned int u32x4;

union Frag { u32x4 u[2]; v16bf v; };

__device__ __forceinline__ unsigned short f2bf(float f) {
  unsigned int u = __float_as_uint(f);
  u += 0x7FFFu + ((u >> 16) & 1u);          // round-to-nearest-even bf16
  return (unsigned short)(u >> 16);
}
__device__ __forceinline__ float sigm(float z) { return 1.f / (1.f + __expf(-z)); }

// ---------------------------------------------------------------------------
// Weight repack: w[co][ci][ky][kx] (f32) -> wpack[co][ (ky*ks+kx)*Cin + ci ] bf16,
// zero padded to KP (mult of 32) in k and CoutPad (mult of 128) in co.
// ---------------------------------------------------------------------------
__global__ void __launch_bounds__(256) repack_k(
    const float* __restrict__ w, unsigned short* __restrict__ wp,
    int cinShift, int Cout, int ks, int KP, int total)
{
  int idx = blockIdx.x * 256 + threadIdx.x;
  if (idx >= total) return;
  int co = idx / KP;
  int k  = idx - co * KP;
  const int Cin  = 1 << cinShift;
  const int Ktot = ks * ks * Cin;
  float v = 0.f;
  if (co < Cout && k < Ktot) {
    int pos = k >> cinShift;
    int ci  = k & (Cin - 1);
    int ky = 0, kx = 0;
    if (ks == 5) { ky = pos / 5; kx = pos - ky * 5; }
    v = w[(((size_t)co * Cin + ci) * ks + ky) * ks + kx];
  }
  wp[idx] = f2bf(v);
}

// ---------------------------------------------------------------------------
// Implicit-GEMM conv via v_wmma_f32_16x16x32_bf16.
// Block = 8 waves = 128 couts x 16 pixels. Patch staged in LDS as bf16 in
// B-fragment layout: half addr = ((k>>5)*16 + n)*32 + (k&31).
// ---------------------------------------------------------------------------
__global__ void __launch_bounds__(256) conv_wmma(
    const float* __restrict__ in, const unsigned short* __restrict__ wpack,
    const float* __restrict__ bias, float* __restrict__ out,
    int cinShift, int Cout, int ks, int pad, int KP)
{
  extern __shared__ unsigned short patch[];
  const int tid  = threadIdx.x;
  const int lane = tid & 31;
  const int wave = tid >> 5;
  const int hi   = lane >> 4;       // half-wave select
  const int ln16 = lane & 15;

  const int nt  = blockIdx.x;       // pixel tile: 256 per batch image
  const int b   = nt >> 8;
  const int rem = nt & 255;
  const int y   = rem >> 2;
  const int x0  = (rem & 3) << 4;

  const int Cin  = 1 << cinShift;
  const int Ktot = ks * ks * Cin;

  // ---- stage input patch (16 pixels x KP k-values) into LDS as bf16 ----
  for (int idx = tid; idx < (KP << 4); idx += 256) {
    const int k = idx >> 4;
    const int n = idx & 15;
    float v = 0.f;
    if (k < Ktot) {
      const int pos = k >> cinShift;
      const int ci  = k & (Cin - 1);
      int ky = 0, kx = 0;
      if (ks == 5) { ky = pos / 5; kx = pos - ky * 5; }
      const int iy = y + ky - pad;
      const int ix = x0 + n + kx - pad;
      if ((unsigned)iy < (unsigned)HH && (unsigned)ix < (unsigned)WW)
        v = in[(((size_t)b * Cin + ci) * HH + iy) * WW + ix];
    }
    patch[(((k >> 5) << 4) + n) * 32 + (k & 31)] = f2bf(v);
  }
  __syncthreads();

  // ---- GEMM: 16 couts (this wave) x 16 pixels, K in chunks of 32 ----
  const int row0 = (blockIdx.y * 8 + wave) * 16;
  const unsigned short* wrow = wpack + (size_t)(row0 + ln16) * KP;
  v8f acc = {0.f,0.f,0.f,0.f,0.f,0.f,0.f,0.f};
  const int nchunks = KP >> 5;
  for (int c = 0; c < nchunks; ++c) {
    Frag a, bm;
    const unsigned short* ap = wrow + c * 32 + hi * 8;   // A: lanes<16 k0..7/16..23
    a.u[0] = *(const u32x4*)(ap);
    a.u[1] = *(const u32x4*)(ap + 16);
    const unsigned short* bp = patch + (((c << 4) + ln16) * 32) + hi * 8;
    bm.u[0] = *(const u32x4*)(bp);                       // B: lanes<16 k0..7/16..23
    bm.u[1] = *(const u32x4*)(bp + 16);
    acc = __builtin_amdgcn_wmma_f32_16x16x32_bf16(false, a.v, false, bm.v,
                                                  (short)0, acc, false, false);
  }

  // ---- store: lane ln16 = pixel, VGPR r -> cout row r + hi*8 ----
  #pragma unroll
  for (int r = 0; r < 8; ++r) {
    const int co = row0 + r + hi * 8;
    if (co < Cout)
      out[(((size_t)b * Cout + co) * HH + y) * WW + x0 + ln16] = acc[r] + bias[co];
  }
}

// ---------------------------------------------------------------------------
// BatchNorm over (B,H,W) per channel, in place. grid = CH blocks.
// ---------------------------------------------------------------------------
__global__ void __launch_bounds__(256) bn_k(float* __restrict__ x,
                                            const float* __restrict__ g,
                                            const float* __restrict__ bt, int CH)
{
  __shared__ float ssum[256], ssq[256];
  __shared__ float mu_s, rs_s;
  const int ch = blockIdx.x, tid = threadIdx.x;
  float s = 0.f, q = 0.f;
  for (int i = tid; i < BB * PIX; i += 256) {
    int b = i >> 12, pix = i & 4095;
    float v = x[((size_t)b * CH + ch) * PIX + pix];
    s += v; q += v * v;
  }
  ssum[tid] = s; ssq[tid] = q; __syncthreads();
  for (int o = 128; o > 0; o >>= 1) {
    if (tid < o) { ssum[tid] += ssum[tid + o]; ssq[tid] += ssq[tid + o]; }
    __syncthreads();
  }
  if (tid == 0) {
    float mu  = ssum[0] * (1.f / (BB * PIX));
    float var = ssq[0] * (1.f / (BB * PIX)) - mu * mu;
    mu_s = mu; rs_s = rsqrtf(var + 1e-5f);
  }
  __syncthreads();
  const float mu = mu_s, rs = rs_s, gg = g[ch], bb = bt[ch];
  for (int i = tid; i < BB * PIX; i += 256) {
    int b = i >> 12, pix = i & 4095;
    size_t a = ((size_t)b * CH + ch) * PIX + pix;
    x[a] = (x[a] - mu) * rs * gg + bb;
  }
}

// ---------------------------------------------------------------------------
// Elementwise kernels (idx over B*64*PIX; b=idx>>18, ch=(idx>>12)&63, pix=idx&4095)
// ---------------------------------------------------------------------------
__global__ void st_gates(const float* __restrict__ t4, const float* __restrict__ s4,
                         const float* __restrict__ x4, float* __restrict__ c,
                         float* __restrict__ m, float* __restrict__ o_buf,
                         float* __restrict__ conc)
{
  int idx = blockIdx.x * blockDim.x + threadIdx.x;
  if (idx >= S64) return;
  int b = idx >> 18, ch = (idx >> 12) & 63, pix = idx & 4095;
  size_t o4 = (((size_t)b << 8) + ch) * PIX + pix;
  float ix = x4[o4], gx = x4[o4+CH64STR], fx = x4[o4+2*CH64STR], ox = x4[o4+3*CH64STR];
  float it = t4[o4], gt = t4[o4+CH64STR], ft = t4[o4+2*CH64STR], ot = t4[o4+3*CH64STR];
  float is = s4[o4], gs = s4[o4+CH64STR], fs = s4[o4+2*CH64STR], os_ = s4[o4+3*CH64STR];
  float i  = sigm(ix + it), i2 = sigm(ix + is);
  float g  = tanhf(gx + gt), g2 = tanhf(gx + gs);
  float f  = sigm(fx + ft + 1.f), f2 = sigm(fx + fs + 1.f);
  float o  = sigm(ox + ot + os_);
  float nc = f * c[idx] + i * g;
  float nm = f2 * m[idx] + i2 * g2;
  c[idx] = nc; m[idx] = nm; o_buf[idx] = o;
  size_t cb = (((size_t)b << 7) + ch) * PIX + pix;
  conc[cb] = nc; conc[cb + CH64STR] = nm;
}

__global__ void mimb_gatesA(const float* __restrict__ t3, const float* __restrict__ s4,
                            const float* __restrict__ x4, float* __restrict__ m,
                            float* __restrict__ ig, float* __restrict__ o_buf)
{
  int idx = blockIdx.x * blockDim.x + threadIdx.x;
  if (idx >= S64) return;
  int b = idx >> 18, ch = (idx >> 12) & 63, pix = idx & 4095;
  size_t o4 = (((size_t)b << 8) + ch) * PIX + pix;
  size_t o3 = ((size_t)b * 192 + ch) * PIX + pix;
  float ix = x4[o4], gx = x4[o4+CH64STR], fx = x4[o4+2*CH64STR], ox = x4[o4+3*CH64STR];
  float is = s4[o4], gs = s4[o4+CH64STR], fs = s4[o4+2*CH64STR], os_ = s4[o4+3*CH64STR];
  float it = t3[o3], gt = t3[o3+CH64STR], ot = t3[o3+2*CH64STR];
  float i  = sigm(ix + it), i2 = sigm(ix + is);
  float g  = tanhf(gx + gt), g2 = tanhf(gx + gs);
  float f2 = sigm(fx + fs + 1.f);
  float o  = sigm(ox + ot + os_);
  m[idx]  = f2 * m[idx] + i2 * g2;
  ig[idx] = i * g;
  o_buf[idx] = o;
}

__global__ void mim_gates(const float* __restrict__ hc, const float* __restrict__ xc,
                          float* __restrict__ c, const float* __restrict__ ct,
                          const float* __restrict__ oc, float* __restrict__ hout)
{
  int idx = blockIdx.x * blockDim.x + threadIdx.x;
  if (idx >= S64) return;
  int b = idx >> 18, ch = (idx >> 12) & 63, pix = idx & 4095;
  size_t o4 = (((size_t)b << 8) + ch) * PIX + pix;
  size_t pp = (size_t)ch * PIX + pix;          // ct/oc broadcast over batch
  float cv = c[idx];
  float ih = hc[o4], gh = hc[o4+CH64STR], fh = hc[o4+2*CH64STR], oh = hc[o4+3*CH64STR];
  float ix = xc[o4], gx = xc[o4+CH64STR], fx = xc[o4+2*CH64STR], ox = xc[o4+3*CH64STR];
  float i_ = sigm(ih + ix + cv * ct[pp]);
  float f_ = sigm(fh + fx + cv * ct[pp + CH64STR] + 1.f);
  float g_ = tanhf(gh + gx);
  float cn = f_ * cv + i_ * g_;
  float o_ = sigm(oh + ox + oc[pp] * cn);
  c[idx] = cn;
  hout[idx] = o_ * tanhf(cn);
}

__global__ void newc_conc(const float* __restrict__ c2, const float* __restrict__ ig,
                          const float* __restrict__ m, float* __restrict__ c,
                          float* __restrict__ conc)
{
  int idx = blockIdx.x * blockDim.x + threadIdx.x;
  if (idx >= S64) return;
  int b = idx >> 18, ch = (idx >> 12) & 63, pix = idx & 4095;
  float nc = c2[idx] + ig[idx];
  c[idx] = nc;
  size_t cb = (((size_t)b << 7) + ch) * PIX + pix;
  conc[cb] = nc; conc[cb + CH64STR] = m[idx];
}

__global__ void hout_k(const float* __restrict__ o_buf, const float* __restrict__ cell,
                       float* __restrict__ h)
{
  int idx = blockIdx.x * blockDim.x + threadIdx.x;
  if (idx < S64) h[idx] = o_buf[idx] * tanhf(cell[idx]);
}

__global__ void diff_k(const float* __restrict__ a, const float* __restrict__ b,
                       float* __restrict__ d)
{
  int idx = blockIdx.x * blockDim.x + threadIdx.x;
  if (idx < S64) d[idx] = a[idx] - b[idx];
}

__global__ void x_make(const float* __restrict__ images, const float* __restrict__ mask,
                       const float* __restrict__ xgen, float* __restrict__ x, int ts)
{
  int idx = blockIdx.x * blockDim.x + threadIdx.x;
  if (idx >= BB * PIX) return;
  int b = idx >> 12, pix = idx & 4095;
  float im = images[((size_t)b * TT + ts) * PIX + pix];
  if (ts >= 5) {
    float mk = mask[((size_t)b * 4 + (ts - 5)) * PIX + pix];
    x[idx] = mk * im + (1.f - mk) * xgen[idx];
  } else {
    x[idx] = im;
  }
}

__global__ void xg_k(const float* __restrict__ h3, const float* __restrict__ w,
                     const float* __restrict__ b, float* __restrict__ dout,
                     float* __restrict__ xgen, int ts)
{
  int idx = blockIdx.x * blockDim.x + threadIdx.x;
  if (idx >= BB * PIX) return;
  int bb = idx >> 12, pix = idx & 4095;
  float s = b[0];
  const float* hp = h3 + (((size_t)bb) << 6) * PIX + pix;
  #pragma unroll 8
  for (int ci = 0; ci < 64; ++ci) s += w[ci] * hp[(size_t)ci * PIX];
  dout[((size_t)bb * 9 + ts) * PIX + pix] = s;
  xgen[idx] = s;
}

// ---------------------------------------------------------------------------
// Host orchestration
// ---------------------------------------------------------------------------
extern "C" void kernel_launch(void* const* d_in, const int* in_sizes, int n_in,
                              void* d_out, int out_size, void* d_ws, size_t ws_size,
                              hipStream_t stream)
{
  (void)in_sizes; (void)n_in; (void)out_size; (void)ws_size;
  const float* images = (const float*)d_in[0];
  const float* mask   = (const float*)d_in[1];
  auto P = [&](int i) { return (const float*)d_in[i]; };
  float* out = (float*)d_out;

  // bump allocator in d_ws
  char* base = (char*)d_ws;
  size_t off = 0;
  auto alloc  = [&](size_t bytes) -> char* {
    char* p = base + off;
    off = (off + bytes + 255) & ~(size_t)255;
    return p;
  };
  auto allocF = [&](size_t n) { return (float*)alloc(n * sizeof(float)); };
  auto allocH = [&](size_t n) { return (unsigned short*)alloc(n * sizeof(unsigned short)); };

  // --- recurrent state (zeroed every call) ---
  float *h[4], *c[4], *hd[3], *cd[3];
  for (int i = 0; i < 4; ++i) h[i]  = allocF(S64);
  for (int i = 0; i < 4; ++i) c[i]  = allocF(S64);
  for (int i = 0; i < 3; ++i) hd[i] = allocF(S64);
  for (int i = 0; i < 3; ++i) cd[i] = allocF(S64);
  float* stm  = allocF(S64);
  float* clc  = allocF(S64);
  float* xgen = allocF(BB * PIX);
  const size_t zeroBytes = off;

  // --- scratch ---
  float* h0old = allocF(S64);
  float* diffb = allocF(S64);
  float* o_buf = allocF(S64);
  float* ig_b  = allocF(S64);
  float* c2_b  = allocF(S64);
  float* cell  = allocF(S64);
  float* tmpA  = allocF((size_t)BB * 256 * PIX);
  float* tmpB  = allocF((size_t)BB * 256 * PIX);
  float* tmpC  = allocF((size_t)BB * 256 * PIX);
  float* conc  = allocF((size_t)BB * 128 * PIX);
  float* xbuf  = allocF(BB * PIX);

  hipMemsetAsync(d_ws, 0, zeroBytes, stream);

  // --- conv descriptors + weight repack (bf16) ---
  struct CD { const float* w; const float* b; int cinShift, Cout, ks, pad, KP, CoutPad;
              unsigned short* wp; };
  auto mkcd = [&](int wi, int bi, int cinShift, int Cout, int ks, int pad) {
    CD d; d.w = P(wi); d.b = P(bi); d.cinShift = cinShift; d.Cout = Cout; d.ks = ks; d.pad = pad;
    int Cin = 1 << cinShift;
    d.KP = ((ks * ks * Cin + 31) / 32) * 32;
    d.CoutPad = ((Cout + 127) / 128) * 128;
    d.wp = allocH((size_t)d.CoutPad * d.KP);
    return d;
  };
  CD STT = mkcd( 2,  3, 6, 256, 5, 2);
  CD STS = mkcd( 4,  5, 6, 256, 5, 2);
  CD STX = mkcd( 6,  7, 0, 256, 5, 2);
  CD STC = mkcd( 8,  9, 7,  64, 1, 0);
  CD MBT = mkcd(16, 17, 6, 192, 5, 2);
  CD MBS = mkcd(18, 19, 6, 256, 5, 2);
  CD MBX = mkcd(20, 21, 6, 256, 5, 2);
  CD MBC = mkcd(22, 23, 7,  64, 1, 0);
  CD MSH = mkcd(30, 31, 6, 256, 5, 2);
  CD MSX = mkcd(32, 33, 6, 256, 5, 2);
  CD MNH = mkcd(40, 41, 6, 256, 5, 2);
  CD MNX = mkcd(42, 43, 6, 256, 5, 2);

  auto repack = [&](const CD& d) {
    int total = d.CoutPad * d.KP;
    repack_k<<<(total + 255) / 256, 256, 0, stream>>>(d.w, d.wp, d.cinShift, d.Cout,
                                                      d.ks, d.KP, total);
  };
  repack(STT); repack(STS); repack(STX); repack(STC);
  repack(MBT); repack(MBS); repack(MBX); repack(MBC);
  repack(MSH); repack(MSX); repack(MNH); repack(MNX);

  auto conv = [&](const CD& d, const float* in, float* o) {
    dim3 grid(NPIX16, d.CoutPad / 128);
    size_t sh = (size_t)(d.KP << 4) * sizeof(unsigned short);
    conv_wmma<<<grid, 256, sh, stream>>>(in, d.wp, d.b, o, d.cinShift, d.Cout,
                                         d.ks, d.pad, d.KP);
  };
  auto bnrm = [&](float* x, int gi, int CH) {
    bn_k<<<CH, 256, 0, stream>>>(x, P(gi), P(gi + 1), CH);
  };

  const int EB = 256, EG = S64 / EB;          // elementwise grid
  const int SG = (BB * PIX + 255) / 256;      // small grid

  for (int ts = 0; ts < TT - 1; ++ts) {
    x_make<<<SG, 256, 0, stream>>>(images, mask, xgen, xbuf, ts);
    hipMemcpyAsync(h0old, h[0], (size_t)S64 * sizeof(float),
                   hipMemcpyDeviceToDevice, stream);

    // ---- ST-LSTM (layer 0) ----
    conv(STT, h[0], tmpA);  bnrm(tmpA, 10, 256);
    conv(STS, stm,  tmpB);  bnrm(tmpB, 12, 256);
    conv(STX, xbuf, tmpC);  bnrm(tmpC, 14, 256);
    st_gates<<<EG, EB, 0, stream>>>(tmpA, tmpB, tmpC, c[0], stm, o_buf, conc);
    conv(STC, conc, cell);
    hout_k<<<EG, EB, 0, stream>>>(o_buf, cell, h[0]);
    diff_k<<<EG, EB, 0, stream>>>(h[0], h0old, diffb);

    // ---- MIM layers 1..3 ----
    for (int i = 1; i < 4; ++i) {
      float* hdi = hd[i - 1];
      float* cdi = cd[i - 1];
      if (ts > 0) {
        const float* inp = (i == 1) ? diffb : hd[i - 2];
        conv(MNH, hdi, tmpA);  bnrm(tmpA, 46, 256);
        conv(MNX, inp, tmpB);  bnrm(tmpB, 48, 256);
        mim_gates<<<EG, EB, 0, stream>>>(tmpA, tmpB, cdi, P(44), P(45), hdi);
      }
      // mimb
      conv(MBT, h[i],     tmpA);  bnrm(tmpA, 24, 192);
      conv(MBS, stm,      tmpB);  bnrm(tmpB, 26, 256);
      conv(MBX, h[i - 1], tmpC);  bnrm(tmpC, 28, 256);
      mimb_gatesA<<<EG, EB, 0, stream>>>(tmpA, tmpB, tmpC, stm, ig_b, o_buf);
      // mim_cell 'ms' (x = hd[i-1], h = c[i], c = clc)
      conv(MSH, c[i], tmpA);  bnrm(tmpA, 36, 256);
      conv(MSX, hdi,  tmpB);  bnrm(tmpB, 38, 256);
      mim_gates<<<EG, EB, 0, stream>>>(tmpA, tmpB, clc, P(34), P(35), c2_b);
      newc_conc<<<EG, EB, 0, stream>>>(c2_b, ig_b, stm, c[i], conc);
      conv(MBC, conc, cell);
      hout_k<<<EG, EB, 0, stream>>>(o_buf, cell, h[i]);
    }

    xg_k<<<SG, 256, 0, stream>>>(h[3], P(50), P(51), out, xgen, ts);
  }
}
